// CliffordInteraction_16149077033518
// MI455X (gfx1250) — compile-verified, matrix operand order checked by reading, not run on hardware
//
#include <hip/hip_runtime.h>
#include <hip/hip_bf16.h>

typedef __attribute__((ext_vector_type(16))) _Float16 v16h;
typedef __attribute__((ext_vector_type(8)))  _Float16 v8h;
typedef __attribute__((ext_vector_type(8)))  float    v8f;

#define B_  8
#define C_  256
#define H_  128
#define W_  128
#define HW_ 16384            // H*W
#define CHW_ 4194304         // C*H*W
#define INV_CHW 2.384185791015625e-07f
#define GC_ 1024             // S*2*C derived channels

__device__ __forceinline__ float silu_f(float x) { return x / (1.0f + __expf(-x)); }

// Load a 16-half WMMA fragment (A row-major or B col-gathered) for wave32 layout.
// p points at row_base + k0 + hi*8 ; halves [0..7] from p, [8..15] from p+16.
template <typename PT>
__device__ __forceinline__ v16h ld_frag(const PT* p) {
  v8h lo = *(const v8h*)(p);
  v8h hi = *(const v8h*)(p + 16);
  v16h r;
#pragma unroll
  for (int i = 0; i < 8; ++i) { r[i] = lo[i]; r[i + 8] = hi[i]; }
  return r;
}

// ---------------------------------------------------------------------------
// K0a: transpose x (NCHW f32) -> xT[b*HW+pos][c] f16, 16x16 LDS tiles
// ---------------------------------------------------------------------------
__global__ __launch_bounds__(256) void k_transpose_x_f16(
    const float* __restrict__ x, _Float16* __restrict__ xT) {
  __shared__ float tile[16][17];
  unsigned bi = blockIdx.x;
  int pt = bi & 1023;            // HW/16 = 1024 tiles
  int ct = (bi >> 10) & 15;      // C/16 = 16 tiles
  int b  = bi >> 14;
  int tx = threadIdx.x & 15, ty = threadIdx.x >> 4;
  int c   = ct * 16 + ty;
  int pos = pt * 16 + tx;
  tile[ty][tx] = x[((size_t)b * C_ + c) * HW_ + pos];
  __syncthreads();
  // write element (pos = pt*16+ty, c = ct*16+tx)
  xT[((size_t)b * HW_ + pt * 16 + ty) * C_ + ct * 16 + tx] = (_Float16)tile[tx][ty];
}

// ---------------------------------------------------------------------------
// K0b: convert w_det (256x256) and w_fp (256x1024) to f16
// ---------------------------------------------------------------------------
__global__ __launch_bounds__(256) void k_convert_weights(
    const float* __restrict__ wdet, const float* __restrict__ wfp,
    _Float16* __restrict__ wdet16, _Float16* __restrict__ wfp16) {
  int i = blockIdx.x * 256 + threadIdx.x;
  if (i < C_ * C_) {
    wdet16[i] = (_Float16)wdet[i];
  } else {
    int j = i - C_ * C_;
    if (j < C_ * GC_) wfp16[j] = (_Float16)wfp[j];
  }
}

// ---------------------------------------------------------------------------
// K1/K3: depthwise 3x3 SAME conv + per-sample sum/sumsq atomics
// ---------------------------------------------------------------------------
__global__ __launch_bounds__(256) void k_dwconv_stats(
    const float* __restrict__ in, const float* __restrict__ wdw,
    float* __restrict__ out, float* __restrict__ stats) {
  int idx = blockIdx.x * 256 + threadIdx.x;     // < B*C*H*W
  int w = idx & (W_ - 1);
  int h = (idx >> 7) & (H_ - 1);
  int c = (idx >> 14) & (C_ - 1);
  int b = idx >> 22;
  const float* wp = wdw + c * 9;
  const float* base = in + ((size_t)b * C_ + c) * HW_;
  float acc = 0.f;
#pragma unroll
  for (int dy = -1; dy <= 1; ++dy) {
    int hh = h + dy;
    if ((unsigned)hh >= H_) continue;
#pragma unroll
    for (int dx = -1; dx <= 1; ++dx) {
      int ww = w + dx;
      if ((unsigned)ww >= W_) continue;
      acc += wp[(dy + 1) * 3 + (dx + 1)] * base[hh * W_ + ww];
    }
  }
  out[idx] = acc;
  // reduce sum / sumsq: wave shuffle then cross-wave LDS, 2 atomics per block
  float s = acc, q = acc * acc;
#pragma unroll
  for (int off = 16; off > 0; off >>= 1) { s += __shfl_down(s, off); q += __shfl_down(q, off); }
  __shared__ float rs[8], rq[8];
  int wid = threadIdx.x >> 5;
  if ((threadIdx.x & 31) == 0) { rs[wid] = s; rq[wid] = q; }
  __syncthreads();
  if (threadIdx.x == 0) {
    float a = 0.f, bb = 0.f;
#pragma unroll
    for (int i = 0; i < 8; ++i) { a += rs[i]; bb += rq[i]; }
    atomicAdd(&stats[b * 2 + 0], a);
    atomicAdd(&stats[b * 2 + 1], bb);
  }
}

// ---------------------------------------------------------------------------
// K2: y = silu(GN(y)) in place (NCHW)
// ---------------------------------------------------------------------------
__global__ __launch_bounds__(256) void k_gn_silu_inplace(
    float* __restrict__ y, const float* __restrict__ stats,
    const float* __restrict__ gg, const float* __restrict__ gb) {
  int idx = blockIdx.x * 256 + threadIdx.x;
  int c = (idx >> 14) & (C_ - 1);
  int b = idx >> 22;
  float mu  = stats[b * 2 + 0] * INV_CHW;
  float var = stats[b * 2 + 1] * INV_CHW - mu * mu;
  float rs  = rsqrtf(var + 1e-6f);
  float v = (y[idx] - mu) * rs * gg[c] + gb[c];
  y[idx] = silu_f(v);
}

// ---------------------------------------------------------------------------
// K4: z2 = silu(GN2(y2)); write position-major zt[b*HW+pos][c] f32,
//     accumulate post-SiLU stats for the gctx GroupNorm.
// ---------------------------------------------------------------------------
__global__ __launch_bounds__(256) void k_gn_silu_transpose_stats(
    const float* __restrict__ y2, const float* __restrict__ stats2,
    const float* __restrict__ gg, const float* __restrict__ gb,
    float* __restrict__ zt, float* __restrict__ stats_out) {
  __shared__ float tile[16][17];
  unsigned bi = blockIdx.x;
  int pt = bi & 1023;
  int ct = (bi >> 10) & 15;
  int b  = bi >> 14;
  int tx = threadIdx.x & 15, ty = threadIdx.x >> 4;
  int c   = ct * 16 + ty;
  int pos = pt * 16 + tx;
  float mu  = stats2[b * 2 + 0] * INV_CHW;
  float var = stats2[b * 2 + 1] * INV_CHW - mu * mu;
  float rsg = rsqrtf(var + 1e-6f);
  float v = y2[((size_t)b * C_ + c) * HW_ + pos];
  v = silu_f((v - mu) * rsg * gg[c] + gb[c]);
  tile[ty][tx] = v;
  // stats of post-SiLU output
  float s = v, q = v * v;
#pragma unroll
  for (int off = 16; off > 0; off >>= 1) { s += __shfl_down(s, off); q += __shfl_down(q, off); }
  __shared__ float rs_[8], rq_[8];
  int wid = threadIdx.x >> 5;
  if ((threadIdx.x & 31) == 0) { rs_[wid] = s; rq_[wid] = q; }
  __syncthreads();
  if (threadIdx.x == 0) {
    float a = 0.f, bb = 0.f;
#pragma unroll
    for (int i = 0; i < 8; ++i) { a += rs_[i]; bb += rq_[i]; }
    atomicAdd(&stats_out[b * 2 + 0], a);
    atomicAdd(&stats_out[b * 2 + 1], bb);
  }
  zt[((size_t)b * HW_ + pt * 16 + ty) * C_ + ct * 16 + tx] = tile[tx][ty];
}

// ---------------------------------------------------------------------------
// K5: det branch 1x1 conv as WMMA GEMM: zdet[pos][n] = xT[pos][:] . w_det[n][:] + b
//     512 threads = 16 waves, one 16(M)x16(N) tile per wave, K=256 in 8 WMMAs.
//     Accumulates per-sample GN stats of z_det.
// ---------------------------------------------------------------------------
__global__ __launch_bounds__(512) void k_det_gemm(
    const _Float16* __restrict__ xT, const _Float16* __restrict__ wdet16,
    const float* __restrict__ bdet, float* __restrict__ zdet,
    float* __restrict__ stats_det) {
  int wave = threadIdx.x >> 5;
  int lane = threadIdx.x & 31;
  int mt = blockIdx.x;                 // 8192 position tiles
  int b  = mt >> 10;                   // 1024 tiles per sample
  size_t posBase = (size_t)mt * 16;    // global b*HW+pos base
  int n   = wave * 16 + (lane & 15);
  int row = lane & 15;
  int hi  = lane >> 4;
  const _Float16* arow = xT + (posBase + row) * C_ + hi * 8;
  const _Float16* brow = wdet16 + (size_t)n * C_ + hi * 8;
  v8f acc = {};
#pragma unroll
  for (int kt = 0; kt < 8; ++kt) {
    v16h a  = ld_frag(arow + kt * 32);
    v16h bf = ld_frag(brow + kt * 32);
    acc = __builtin_amdgcn_wmma_f32_16x16x32_f16(false, a, false, bf,
                                                 (short)0, acc, false, false);
  }
  float bias = bdet[n];
  float s = 0.f, q = 0.f;
#pragma unroll
  for (int r = 0; r < 8; ++r) {
    float v = acc[r] + bias;
    int pos = hi * 8 + r;                       // M index in tile
    zdet[(posBase + pos) * C_ + n] = v;
    s += v; q += v * v;
  }
#pragma unroll
  for (int off = 16; off > 0; off >>= 1) { s += __shfl_down(s, off); q += __shfl_down(q, off); }
  __shared__ float rs_[16], rq_[16];
  if (lane == 0) { rs_[wave] = s; rq_[wave] = q; }
  __syncthreads();
  if (threadIdx.x == 0) {
    float a = 0.f, bb = 0.f;
#pragma unroll
    for (int i = 0; i < 16; ++i) { a += rs_[i]; bb += rq_[i]; }
    atomicAdd(&stats_det[b * 2 + 0], a);
    atomicAdd(&stats_det[b * 2 + 1], bb);
  }
}

// ---------------------------------------------------------------------------
// K6: fusion. Per block: 16 positions. Normalize z_det/z_ctx into LDS, build the
//     16x1024 f16 g-tile (dot=silu(prod), wedge), then 16 waves x 32 WMMAs
//     against w_fp (f16) and store NCHW output.
// ---------------------------------------------------------------------------
__global__ __launch_bounds__(512) void k_fusion(
    const float* __restrict__ zt_ctx, const float* __restrict__ zdet,
    const float* __restrict__ stats_ctx, const float* __restrict__ stats_det,
    const float* __restrict__ gcg, const float* __restrict__ gcb,
    const float* __restrict__ gdg, const float* __restrict__ gdb,
    const _Float16* __restrict__ wfp16, const float* __restrict__ bfp,
    float* __restrict__ out) {
  __shared__ float    s_zd[16][C_];
  __shared__ float    s_zc[16][C_];
  __shared__ _Float16 s_g [16][GC_];

  int mt = blockIdx.x;
  int b  = mt >> 10;
  size_t posBase = (size_t)mt * 16;
  int posIn = (mt & 1023) * 16;        // position within sample

  float muc  = stats_ctx[b * 2 + 0] * INV_CHW;
  float varc = stats_ctx[b * 2 + 1] * INV_CHW - muc * muc;
  float rsc  = rsqrtf(varc + 1e-6f);
  float mud  = stats_det[b * 2 + 0] * INV_CHW;
  float vard = stats_det[b * 2 + 1] * INV_CHW - mud * mud;
  float rsd  = rsqrtf(vard + 1e-6f);

  // stage + normalize (16*256 contiguous elements per array)
  for (int i = threadIdx.x; i < 16 * C_; i += 512) {
    int p = i >> 8, c = i & (C_ - 1);
    float vd = zdet[(posBase + p) * C_ + c];
    float vc = zt_ctx[(posBase + p) * C_ + c];
    s_zd[p][c] = (vd - mud) * rsd * gdg[c] + gdb[c];
    s_zc[p][c] = (vc - muc) * rsc * gcg[c] + gcb[c];
  }
  __syncthreads();

  // derived channels: ch = (si*2 + pp)*256 + c, shift = si+1
  for (int i = threadIdx.x; i < 16 * GC_; i += 512) {
    int p  = i >> 10;
    int ch = i & (GC_ - 1);
    int c  = ch & (C_ - 1);
    int pp = (ch >> 8) & 1;
    int si = ch >> 9;
    int cm = (c - (si + 1)) & (C_ - 1);
    float zd  = s_zd[p][c],  zc  = s_zc[p][c];
    float zdm = s_zd[p][cm], zcm = s_zc[p][cm];
    float prod = zd * zcm;
    float gv = (pp == 0) ? silu_f(prod) : (prod - zdm * zc);
    s_g[p][ch] = (_Float16)gv;
  }
  __syncthreads();

  // GEMM: out[pos][n] = g[pos][:] . w_fp[n][:] + b_fp[n]
  int wave = threadIdx.x >> 5;
  int lane = threadIdx.x & 31;
  int n   = wave * 16 + (lane & 15);
  int row = lane & 15;
  int hi  = lane >> 4;
  const _Float16* brow = wfp16 + (size_t)n * GC_ + hi * 8;
  v8f acc = {};
#pragma unroll 8
  for (int kt = 0; kt < 32; ++kt) {
    v16h a  = ld_frag(&s_g[row][kt * 32 + hi * 8]);
    v16h bf = ld_frag(brow + kt * 32);
    acc = __builtin_amdgcn_wmma_f32_16x16x32_f16(false, a, false, bf,
                                                 (short)0, acc, false, false);
  }
  float bias = bfp[n];
  size_t obase = ((size_t)b * C_ + n) * HW_ + posIn + hi * 8;
  float4 v0 = make_float4(acc[0] + bias, acc[1] + bias, acc[2] + bias, acc[3] + bias);
  float4 v1 = make_float4(acc[4] + bias, acc[5] + bias, acc[6] + bias, acc[7] + bias);
  *(float4*)(out + obase)     = v0;
  *(float4*)(out + obase + 4) = v1;
}

// ---------------------------------------------------------------------------
extern "C" void kernel_launch(void* const* d_in, const int* in_sizes, int n_in,
                              void* d_out, int out_size, void* d_ws, size_t ws_size,
                              hipStream_t stream) {
  const float* x      = (const float*)d_in[0];
  const float* w_dw1  = (const float*)d_in[1];
  const float* gn1_g  = (const float*)d_in[2];
  const float* gn1_b  = (const float*)d_in[3];
  const float* w_dw2  = (const float*)d_in[4];
  const float* gn2_g  = (const float*)d_in[5];
  const float* gn2_b  = (const float*)d_in[6];
  const float* w_det  = (const float*)d_in[7];
  const float* b_det  = (const float*)d_in[8];
  const float* gctx_g = (const float*)d_in[9];
  const float* gctx_b = (const float*)d_in[10];
  const float* gdet_g = (const float*)d_in[11];
  const float* gdet_b = (const float*)d_in[12];
  const float* w_fp   = (const float*)d_in[13];
  const float* b_fp   = (const float*)d_in[14];
  float* out = (float*)d_out;

  char* ws = (char*)d_ws;
  const size_t N = (size_t)B_ * CHW_;                 // 33.5M elements
  size_t OFF_XT = 0;                                  // f16 transposed x
  size_t OFF_A  = OFF_XT + N * sizeof(_Float16);      // fp32 buf A
  size_t OFF_B  = OFF_A + N * sizeof(float);          // fp32 buf B
  size_t OFF_WD = OFF_B + N * sizeof(float);
  size_t OFF_WF = OFF_WD + (size_t)C_ * C_ * sizeof(_Float16);
  size_t OFF_ST = OFF_WF + (size_t)C_ * GC_ * sizeof(_Float16);

  _Float16* xT     = (_Float16*)(ws + OFF_XT);
  float*    bufA   = (float*)(ws + OFF_A);
  float*    bufB   = (float*)(ws + OFF_B);
  _Float16* wdet16 = (_Float16*)(ws + OFF_WD);
  _Float16* wfp16  = (_Float16*)(ws + OFF_WF);
  float*    stats  = (float*)(ws + OFF_ST);           // 4 arrays of B*2 floats
  float* stats1   = stats;
  float* stats2   = stats + 16;
  float* statsCtx = stats + 32;
  float* statsDet = stats + 48;

  hipMemsetAsync(stats, 0, 64 * sizeof(float), stream);

  const int EBLK = (int)(N / 256);                    // 131072 blocks, elementwise grids
  k_transpose_x_f16<<<EBLK, 256, 0, stream>>>(x, xT);
  k_convert_weights<<<(C_ * C_ + C_ * GC_ + 255) / 256, 256, 0, stream>>>(
      w_det, w_fp, wdet16, wfp16);

  k_dwconv_stats<<<EBLK, 256, 0, stream>>>(x, w_dw1, bufA, stats1);
  k_gn_silu_inplace<<<EBLK, 256, 0, stream>>>(bufA, stats1, gn1_g, gn1_b);
  k_dwconv_stats<<<EBLK, 256, 0, stream>>>(bufA, w_dw2, bufB, stats2);
  k_gn_silu_transpose_stats<<<EBLK, 256, 0, stream>>>(
      bufB, stats2, gn2_g, gn2_b, bufA, statsCtx);     // bufA <- z_ctx (pos-major)

  k_det_gemm<<<(int)(N / C_ / 16), 512, 0, stream>>>(  // 8192 tiles
      xT, wdet16, b_det, bufB, statsDet);              // bufB <- z_det (pos-major)

  k_fusion<<<(int)(N / C_ / 16), 512, 0, stream>>>(
      bufA, bufB, statsCtx, statsDet,
      gctx_g, gctx_b, gdet_g, gdet_b, wfp16, b_fp, out);
}
